// SinkhornOTLoss_7447473291840
// MI455X (gfx1250) — matrix operand order repaired and checked by reading, not compiled, not run on hardware
//
#include <hip/hip_runtime.h>
#include <hip/hip_bf16.h>

typedef __attribute__((ext_vector_type(16))) _Float16 v16h;
typedef __attribute__((ext_vector_type(8)))  _Float16 v8h;
typedef __attribute__((ext_vector_type(4)))  _Float16 v4h;
typedef __attribute__((ext_vector_type(8)))  float    v8f;

#define INV_EPS  10.0f      // 1/0.1
#define EPS_DIV  1e-8f
#define NN       256
#define TPB      256        // 8 wave32s, 2 output blocks per wave
#define ITERS    100

union V16 { v16h v; v8h p[2]; };

// One Sinkhorn half-step: two 16-row matvec blocks per wave via WMMA, then the
// diagonal scaling fused in-register. All pointer args are loop-invariant, so
// inside the unrolled loop every ds_load_b128 is base-VGPR + immediate offset.
__device__ __forceinline__ void sink_step(const _Float16* __restrict__ Wb0,
                                          const _Float16* __restrict__ Wb1,
                                          const _Float16* __restrict__ vecA,
                                          int hi, int oidx, float scale,
                                          _Float16* __restrict__ dst,
                                          float* __restrict__ dstf, bool last) {
  v8f acc0 = {}, acc1 = {};
#pragma unroll
  for (int kc = 0; kc < 8; ++kc) {
    const int nc = kc * 32;
    V16 A, B0, B1;
    A.p[0]  = *(const v8h*)(vecA + nc);          // ds_load_b128 (imm offset)
    A.p[1]  = *(const v8h*)(vecA + nc + 8);
    B0.p[0] = *(const v8h*)(Wb0 + nc);
    B0.p[1] = *(const v8h*)(Wb0 + nc + 8);
    B1.p[0] = *(const v8h*)(Wb1 + nc);
    B1.p[1] = *(const v8h*)(Wb1 + nc + 8);
    acc0 = __builtin_amdgcn_wmma_f32_16x16x32_f16(false, A.v, false, B0.v,
                                                  (short)0, acc0, false, false);
    acc1 = __builtin_amdgcn_wmma_f32_16x16x32_f16(false, A.v, false, B1.v,
                                                  (short)0, acc1, false, false);
  }
  // rows of D are replicated: lanes 0-15 own block0 (acc0 row0, VGPR0),
  // lanes 16-31 own block1 (acc1 row8, also VGPR0) -> branch-free epilogue.
  const float y = (hi ? acc1[0] : acc0[0]) + EPS_DIV;
  const float r = scale * __builtin_amdgcn_rcpf(y);   // fast v_rcp_f32
  dst[oidx] = (_Float16)r;
  if (last) dstf[oidx] = r;                           // f32 copy for phase 3
}

__global__ void __launch_bounds__(TPB)
sinkhorn_ot_kernel(const float* __restrict__ cost,
                   const float* __restrict__ mp,
                   const float* __restrict__ mt,
                   float* __restrict__ out, float inv_B) {
  extern __shared__ char smem[];
  _Float16* Kh = (_Float16*)smem;            // K  row-major [n][m], 128 KB
  _Float16* Kt = Kh + NN * NN;               // K^T row-major [m][n], 128 KB
  float* uf  = (float*)(smem + 2 * NN * NN * sizeof(_Float16));
  float* vf  = uf + NN;
  float* af  = vf + NN;
  float* bf  = af + NN;
  float* red = bf + NN;                      // 256-entry reduction scratch
  _Float16* uh = (_Float16*)(red + TPB);
  _Float16* vh = uh + NN;

  const int b = blockIdx.x;
  const int t = threadIdx.x;
  const float* cb = cost + (size_t)b * NN * NN;

  // ---- Phase 1: K = exp(-cost/eps) -> f16, store row-major + transposed ----
  for (int base = t * 4; base < NN * NN; base += TPB * 4) {
    const float4 c4 = *(const float4*)(cb + base);           // global_load_b128
    const int n = base >> 8, m = base & 255;
    _Float16 k0 = (_Float16)__expf(-c4.x * INV_EPS);
    _Float16 k1 = (_Float16)__expf(-c4.y * INV_EPS);
    _Float16 k2 = (_Float16)__expf(-c4.z * INV_EPS);
    _Float16 k3 = (_Float16)__expf(-c4.w * INV_EPS);
    v4h pack = {k0, k1, k2, k3};
    *(v4h*)(Kh + base) = pack;                               // ds_store_b64
    Kt[(m + 0) * NN + n] = k0;                               // transposed scatter
    Kt[(m + 1) * NN + n] = k1;
    Kt[(m + 2) * NN + n] = k2;
    Kt[(m + 3) * NN + n] = k3;
  }

  // ---- mass sums: two sequential 256-way tree reductions ----
  red[t] = mp[b * NN + t];
  __syncthreads();
  for (int s = 128; s > 0; s >>= 1) {
    if (t < s) red[t] += red[t + s];
    __syncthreads();
  }
  const float sa = red[0] + EPS_DIV;
  __syncthreads();
  red[t] = mt[b * NN + t];
  __syncthreads();
  for (int s = 128; s > 0; s >>= 1) {
    if (t < s) red[t] += red[t + s];
    __syncthreads();
  }
  const float sb = red[0] + EPS_DIV;
  __syncthreads();

  af[t] = mp[b * NN + t] / sa;
  bf[t] = mt[b * NN + t] / sb;
  uh[t] = (_Float16)1.0f;
  __syncthreads();

  // ---- Phase 2: 100 Sinkhorn iterations, fully in LDS via WMMA ----
  // Wave w computes output blocks 2w (lanes 0-15) and 2w+1 (lanes 16-31).
  // A operand as loaded here (contiguous k, not AMD's table order): lanes 0-15 hold K {0..15}; lanes 16-31
  // hold {16..31}. B layout (32x16): lane holds a contiguous row
  // segment of W; k-half selected by lane/16.
  const int wv = t >> 5, ln = t & 31;
  const int hi = (ln & 16) ? 1 : 0;
  const int cl = ln & 15;
  const int oidx = wv * 32 + ln;             // element owned by this lane
  const float aval = af[oidx];
  const float bval = bf[oidx];
  const int brow0 = (wv * 32 + cl) * NN + (hi ? 16 : 0);
  const int brow1 = (wv * 32 + 16 + cl) * NN + (hi ? 16 : 0);
  const int aoff = hi ? 16 : 0;
  // loop-invariant operand base pointers
  const _Float16* Kt0 = Kt + brow0;
  const _Float16* Kt1 = Kt + brow1;
  const _Float16* Kh0 = Kh + brow0;
  const _Float16* Kh1 = Kh + brow1;
  const _Float16* uhA = uh + aoff;
  const _Float16* vhA = vh + aoff;

#pragma unroll 1
  for (int it = 0; it < ITERS; ++it) {
    const bool last = (it == ITERS - 1);
    // y[m] = sum_n K[n,m] u[n] -> W = K^T ; v = b/(y+eps)
    sink_step(Kt0, Kt1, uhA, hi, oidx, bval, vh, vf, last);
    __syncthreads();
    // x[n] = sum_m K[n,m] v[m] -> W = K ; u = a/(x+eps)
    sink_step(Kh0, Kh1, vhA, hi, oidx, aval, uh, uf, last);
    __syncthreads();
  }

  // ---- Phase 3: ot = sum u[n] * K[n,m] * v[m] * C[n,m]  (fresh f32 K) ----
  float loc = 0.0f;
  for (int base = t * 4; base < NN * NN; base += TPB * 4) {
    const float4 c4 = *(const float4*)(cb + base);
    const int n = base >> 8, m = base & 255;
    const float un = uf[n];
    loc += un * vf[m + 0] * c4.x * __expf(-c4.x * INV_EPS);
    loc += un * vf[m + 1] * c4.y * __expf(-c4.y * INV_EPS);
    loc += un * vf[m + 2] * c4.z * __expf(-c4.z * INV_EPS);
    loc += un * vf[m + 3] * c4.w * __expf(-c4.w * INV_EPS);
  }
  red[t] = loc;
  __syncthreads();
  for (int s = TPB / 2; s > 0; s >>= 1) {
    if (t < s) red[t] += red[t + s];
    __syncthreads();
  }
  if (t == 0) atomicAdd(out, red[0] * inv_B);
}

__global__ void zero_out_kernel(float* out) { out[0] = 0.0f; }

extern "C" void kernel_launch(void* const* d_in, const int* in_sizes, int n_in,
                              void* d_out, int out_size, void* d_ws, size_t ws_size,
                              hipStream_t stream) {
  const float* cost = (const float*)d_in[0];   // [B, 256, 256] f32
  const float* mp   = (const float*)d_in[1];   // [B, 256] f32
  const float* mt   = (const float*)d_in[2];   // [B, 256] f32
  float* out = (float*)d_out;                  // scalar f32

  const int B = in_sizes[1] / NN;              // 1024

  // LDS: 2 * 128KB f16 matrices + vectors + reduction scratch
  const size_t shmem = 2 * NN * NN * sizeof(_Float16)
                     + (4 * NN + TPB) * sizeof(float)
                     + 2 * NN * sizeof(_Float16);

  zero_out_kernel<<<1, 1, 0, stream>>>(out);
  sinkhorn_ot_kernel<<<B, TPB, shmem, stream>>>(cost, mp, mt, out, 1.0f / (float)B);
}
